// CRF_26637387170323
// MI455X (gfx1250) — matrix operand order repaired and checked by reading, not compiled, not run on hardware
//
#include <hip/hip_runtime.h>
#include <hip/hip_bf16.h>
#include <cstddef>

typedef __attribute__((ext_vector_type(16))) _Float16 v16h;
typedef __attribute__((ext_vector_type(8)))  _Float16 v8h;
typedef __attribute__((ext_vector_type(8)))  float    v8f;

#define CRF_B 512
#define CRF_T 512
#define CRF_K 128
#define MB    16             // batches per block == WMMA M
#define PSTR  (CRF_K + 8)    // padded LDS stride (halves) for the P panel
#define SSTR  (CRF_K + 4)    // padded LDS stride (floats) for the SNew panel
#define NPART (CRF_B + CRF_B / MB)   // 512 score + 32 forward partials

#define LOG2E 1.4426950408889634f
#define LN2   0.6931471805599453f

// ---------------------------------------------------------------------------
// Prep: Et[j][i] = exp(trans[i][j])  (f16, row-major over j so a B-operand
// lane (column j of E) reads contiguous halves)
// ---------------------------------------------------------------------------
__global__ void crf_prep_kernel(const float* __restrict__ trans,
                                _Float16* __restrict__ Et) {
  int idx = blockIdx.x * blockDim.x + threadIdx.x;
  if (idx < CRF_K * CRF_K) {
    int j = idx >> 7, i = idx & (CRF_K - 1);
    Et[j * CRF_K + i] = (_Float16)__expf(trans[i * CRF_K + j]);
  }
}

// ---------------------------------------------------------------------------
// Numerator: one block per batch, threads parallel over time steps.
// Sum masked (trans + emission) terms; max-reduce the last masked index.
// ---------------------------------------------------------------------------
__global__ __launch_bounds__(256) void crf_score_kernel(
    const float* __restrict__ em, const int* __restrict__ tags,
    const unsigned char* __restrict__ mask,
    const float* __restrict__ start, const float* __restrict__ endt,
    const float* __restrict__ trans, float* __restrict__ partial) {
  __shared__ float red[256];
  __shared__ int   lastred[256];
  const int b   = blockIdx.x;
  const int tid = threadIdx.x;

  float sc = 0.f;
  int lastIdx = 0;
  for (int t = 1 + tid; t < CRF_T; t += 256) {
    int cur  = tags[b * CRF_T + t];
    int prev = tags[b * CRF_T + t - 1];
    if (mask[b * CRF_T + t]) {
      sc += trans[prev * CRF_K + cur] +
            em[((size_t)b * CRF_T + t) * CRF_K + cur];
      lastIdx = t;
    }
  }
  red[tid] = sc;
  lastred[tid] = lastIdx;
  __syncthreads();
  for (int s = 128; s > 0; s >>= 1) {
    if (tid < s) {
      red[tid] += red[tid + s];
      lastred[tid] = max(lastred[tid], lastred[tid + s]);
    }
    __syncthreads();
  }
  if (tid == 0) {
    int tg0 = tags[b * CRF_T];
    int lastTag = tags[b * CRF_T + lastred[0]];
    partial[b] = red[0] + start[tg0] + em[(size_t)b * CRF_T * CRF_K + tg0] +
                 endt[lastTag];
  }
}

// ---------------------------------------------------------------------------
// Persistent forward recursion. One block owns a 16-batch alpha panel kept
// entirely in registers (8 floats per thread, (r,c) tiling). Per step
// (2 barriers total):
//   A: row-max via 4-step __shfl_xor butterfly (row = one 16-lane half-wave)
//   C: P = exp2((alpha-max)*log2e) f16 -> LDS panel      | barrier
//   D: stage all 8 A-frag ds_load_b128 first, then a 4x
//      v_wmma_f32_16x16x32_f16 accumulation chain (graduated dscnt waits);
//      SNew = rowmax + log(acc) -> LDS                   | barrier
//   E: branch-free masked update with em(t) loads hoisted to phase A and
//      global_prefetch of em(t+2)  (the only HBM stream).
// ---------------------------------------------------------------------------
__global__ __launch_bounds__(256) void crf_forward_kernel(
    const float* __restrict__ em, const unsigned char* __restrict__ mask,
    const float* __restrict__ start, const float* __restrict__ endt,
    const _Float16* __restrict__ Et, float* __restrict__ partial) {
  __shared__ __align__(16) _Float16 Psh[MB][PSTR];        // ~4.3 KB
  __shared__ __align__(16) float    SNew[MB * SSTR];      // ~8.3 KB
  __shared__ __align__(16) float    rowstat[MB];
  __shared__ unsigned char          mskSh[MB][CRF_T];     // 8 KB

  const int tid    = threadIdx.x;
  const int b0     = blockIdx.x * MB;
  const int wv     = tid >> 5;        // wave id 0..7 -> j-tile
  const int lane   = tid & 31;
  const int nIdx   = lane & 15;       // column within tile (B/D operand lane)
  const int hiHalf = lane >> 4;       // lane group 0/1 per WMMA VGPR layout
  const int r      = tid >> 4;        // row 0..15 for elementwise phases
  const int c      = tid & 15;        // 8-col strip per thread

  // ---- one-time: mask panel to LDS ----
  for (int idx = tid; idx < MB * CRF_T; idx += 256)
    mskSh[idx >> 9][idx & (CRF_T - 1)] =
        mask[(size_t)(b0 + (idx >> 9)) * CRF_T + (idx & (CRF_T - 1))];

  // ---- one-time: alpha0 = start + em[:,0,:] into registers ----
  float a[8];
  const float* emRow = &em[((size_t)(b0 + r) * CRF_T) * CRF_K + c * 8];
  {
    const float4* ep = (const float4*)emRow;
    const float4* sp = (const float4*)&start[c * 8];
    float4 e0 = ep[0], e1 = ep[1], s0 = sp[0], s1 = sp[1];
    a[0] = s0.x + e0.x; a[1] = s0.y + e0.y; a[2] = s0.z + e0.z; a[3] = s0.w + e0.w;
    a[4] = s1.x + e1.x; a[5] = s1.y + e1.y; a[6] = s1.z + e1.z; a[7] = s1.w + e1.w;
  }

  // ---- one-time: constant B fragments of E for this wave's j-tile.
  // B (32x16) fragment k order used here (same as A): lanes 0-15 hold N=lane, K=kc*32+0..15;
  // lanes 16-31 hold N=lane-16, K=kc*32+16..31 (contiguous in Et row).
  v16h Bfrag[4];
  {
    const _Float16* brow = Et + (size_t)(wv * 16 + nIdx) * CRF_K;
    #pragma unroll
    for (int kc = 0; kc < 4; ++kc)
      Bfrag[kc] = *(const v16h*)(brow + kc * 32 + hiHalf * 16);
  }
  __syncthreads();

  for (int t = 1; t < CRF_T; ++t) {
    // ---- hoisted emission stream for phase E (covers exp+WMMA latency) ----
    const float4* ep = (const float4*)(emRow + (size_t)t * CRF_K);
    float4 e0 = ep[0], e1 = ep[1];
    bool on = mskSh[r][t] != 0;
    {   // prefetch the stream two steps ahead
      int tp = (t + 2 < CRF_T) ? t + 2 : CRF_T - 1;
      __builtin_prefetch(emRow + (size_t)tp * CRF_K, 0, 0);
    }

    // ---- A: per-row max, 16-lane butterfly (stays within half-wave) ----
    float mx = a[0];
    #pragma unroll
    for (int e = 1; e < 8; ++e) mx = fmaxf(mx, a[e]);
    #pragma unroll
    for (int k = 8; k >= 1; k >>= 1)
      mx = fmaxf(mx, __shfl_xor(mx, k, 32));
    if (c == 0) rowstat[r] = mx;      // for WMMA-layout phase D

    // ---- C: P = exp2((alpha - rowmax) * log2e), f16 ----
    #pragma unroll
    for (int e = 0; e < 8; ++e)
      Psh[r][c * 8 + e] =
          (_Float16)__builtin_amdgcn_exp2f((a[e] - mx) * LOG2E);
    __syncthreads();

    // ---- D: acc(16x16) += P(16x32) x E(32x16), 4 k-chunks ----
    // A 16x32 fragment k order used here (same as B): lanes 0-15 hold M=lane, K=kc*32+0..15;
    // lanes 16-31 hold M=lane-16, K=kc*32+16..31.
    // Stage ALL fragment loads first so dscnt waits are graduated instead of
    // a full drain before each WMMA.
    v16h aFrag[4];
    {
      const _Float16* prow = &Psh[nIdx][0];
      #pragma unroll
      for (int kc = 0; kc < 4; ++kc) {
        v8h a0 = *(const v8h*)(prow + kc * 32 + hiHalf * 16);
        v8h a1 = *(const v8h*)(prow + kc * 32 + hiHalf * 16 + 8);
        v16h aa;
        #pragma unroll
        for (int e = 0; e < 8; ++e) { aa[e] = a0[e]; aa[e + 8] = a1[e]; }
        aFrag[kc] = aa;
      }
    }
    v8f acc = {};
    #pragma unroll
    for (int kc = 0; kc < 4; ++kc)
      acc = __builtin_amdgcn_wmma_f32_16x16x32_f16(
          false, aFrag[kc], false, Bfrag[kc], (short)0, acc, false, false);

    // D layout: lanes 0-15 -> M=vgpr, lanes 16-31 -> M=vgpr+8; N=lane&15.
    // SNew[m][j] = rowmax[m] + log(acc)  (acc in [0.9, 141]: raw v_log ok)
    {
      const float4* rs = (const float4*)&rowstat[hiHalf * 8];
      float4 r0 = rs[0], r1 = rs[1];
      float rsv[8] = {r0.x, r0.y, r0.z, r0.w, r1.x, r1.y, r1.z, r1.w};
      int j = (wv << 4) + nIdx;
      #pragma unroll
      for (int rr = 0; rr < 8; ++rr)
        SNew[(hiHalf * 8 + rr) * SSTR + j] =
            rsv[rr] + __builtin_amdgcn_logf(acc[rr]) * LN2;
    }
    __syncthreads();

    // ---- E: branch-free masked update (em already in registers) ----
    {
      const float4* sp = (const float4*)&SNew[r * SSTR + c * 8];
      float4 s0 = sp[0], s1 = sp[1];
      float ev[8] = {e0.x, e0.y, e0.z, e0.w, e1.x, e1.y, e1.z, e1.w};
      float sv[8] = {s0.x, s0.y, s0.z, s0.w, s1.x, s1.y, s1.z, s1.w};
      #pragma unroll
      for (int e = 0; e < 8; ++e) a[e] = on ? (sv[e] + ev[e]) : a[e];
    }
    // no barrier: next phase A uses only this thread's registers; every
    // LDS buffer rewrite is separated from its readers by >=1 barrier.
  }

  // ---- denom_b = logsumexp_j(alpha[b,j] + end[j]) ; partial = -sum ----
  float ed[8];
  {
    const float4* epd = (const float4*)&endt[c * 8];
    float4 d0 = epd[0], d1 = epd[1];
    ed[0] = d0.x; ed[1] = d0.y; ed[2] = d0.z; ed[3] = d0.w;
    ed[4] = d1.x; ed[5] = d1.y; ed[6] = d1.z; ed[7] = d1.w;
  }
  float mx = a[0] + ed[0];
  #pragma unroll
  for (int e = 1; e < 8; ++e) mx = fmaxf(mx, a[e] + ed[e]);
  #pragma unroll
  for (int k = 8; k >= 1; k >>= 1)
    mx = fmaxf(mx, __shfl_xor(mx, k, 32));
  float sm = 0.f;
  #pragma unroll
  for (int e = 0; e < 8; ++e)
    sm += __builtin_amdgcn_exp2f((a[e] + ed[e] - mx) * LOG2E);
  #pragma unroll
  for (int k = 8; k >= 1; k >>= 1)
    sm += __shfl_xor(sm, k, 32);
  if (c == 0) rowstat[r] = mx + __builtin_amdgcn_logf(sm) * LN2;
  __syncthreads();
  if (tid == 0) {
    float tot = 0.f;
    for (int m = 0; m < MB; ++m) tot += rowstat[m];
    partial[CRF_B + blockIdx.x] = -tot;  // negated denominator contribution
  }
}

// ---------------------------------------------------------------------------
// Deterministic final reduction over 544 partials (fixed-shape tree).
// ---------------------------------------------------------------------------
__global__ __launch_bounds__(256) void crf_finalize_kernel(
    const float* __restrict__ partial, float* __restrict__ out) {
  __shared__ float red[256];
  const int tid = threadIdx.x;
  float s = 0.f;
  for (int i = tid; i < NPART; i += 256) s += partial[i];
  red[tid] = s;
  __syncthreads();
  for (int k = 128; k > 0; k >>= 1) {
    if (tid < k) red[tid] += red[tid + k];
    __syncthreads();
  }
  if (tid == 0) out[0] = red[0];
}

extern "C" void kernel_launch(void* const* d_in, const int* in_sizes, int n_in,
                              void* d_out, int out_size, void* d_ws, size_t ws_size,
                              hipStream_t stream) {
  (void)in_sizes; (void)n_in; (void)out_size; (void)ws_size;
  const float*         em    = (const float*)d_in[0];
  const int*           tags  = (const int*)d_in[1];
  const unsigned char* mask  = (const unsigned char*)d_in[2];
  const float*         start = (const float*)d_in[3];
  const float*         endt  = (const float*)d_in[4];
  const float*         trans = (const float*)d_in[5];

  _Float16* Et      = (_Float16*)d_ws;                                  // 32 KB
  float*    partial = (float*)((char*)d_ws + CRF_K * CRF_K * sizeof(_Float16));

  crf_prep_kernel<<<(CRF_K * CRF_K + 255) / 256, 256, 0, stream>>>(trans, Et);
  crf_score_kernel<<<CRF_B, 256, 0, stream>>>(em, tags, mask, start, endt,
                                              trans, partial);
  crf_forward_kernel<<<CRF_B / MB, 256, 0, stream>>>(em, mask, start, endt, Et,
                                                     partial);
  crf_finalize_kernel<<<1, 256, 0, stream>>>(partial, (float*)d_out);
}
